// EGNNEncoder_2130303778889
// MI455X (gfx1250) — compile-verified, matrix operand order checked by reading, not run on hardware
//
#include <hip/hip_runtime.h>

typedef __attribute__((ext_vector_type(16))) _Float16 v16h;
typedef __attribute__((ext_vector_type(8)))  _Float16 v8h;
typedef __attribute__((ext_vector_type(8)))  float    v8f;
typedef __attribute__((ext_vector_type(4)))  float    v4f;

#define Bsz   8
#define Nn    128
#define Dd    128
#define Mm    16
#define DEPTH 4
#define EIN   257
#define H1    514
#define KP    544      // H1 padded to 17*32
#define NT1   34       // KP/16 column tiles for P/Q GEMM
#define ROWS  1024     // B*N

__device__ __forceinline__ float siluf(float x) {
    return x / (1.0f + __expf(-x));
}

// fragment K-offset: element e (0..15), lane (0..31)
__device__ __forceinline__ int kofs(int e, int lane) {
    return (e < 8 ? e : e + 8) + ((lane & 16) ? 8 : 0);
}

// ---------------- prep kernels: swizzle weights into WMMA B-fragment layout ----------------

// e_w1 halves -> frags: [l][half][dchunk(4)][ntile(34)][lane*16+e]
__global__ void k_prep_w1frag(const float* __restrict__ e_w1, _Float16* __restrict__ frag) {
    int idx = blockIdx.x * 256 + threadIdx.x;
    if (idx >= 4 * 2 * 4 * 34 * 512) return;
    int e = idx & 15, lane = (idx >> 4) & 31, t = idx >> 9;
    int nt = t % 34; t /= 34;
    int dc = t % 4;  t /= 4;
    int half = t & 1, l = t >> 1;
    int drow = half * 128 + dc * 32 + kofs(e, lane);
    int col  = nt * 16 + (lane & 15);
    float v = (col < H1) ? e_w1[(l * EIN + drow) * H1 + col] : 0.f;
    frag[idx] = (_Float16)v;
}

// e_w2 -> frags: [l][kchunk(17)][lane*16+e]
__global__ void k_prep_ew2frag(const float* __restrict__ e_w2, _Float16* __restrict__ frag) {
    int idx = blockIdx.x * 256 + threadIdx.x;
    if (idx >= 4 * 17 * 512) return;
    int e = idx & 15, lane = (idx >> 4) & 31, t = idx >> 9;
    int kc = t % 17, l = t / 17;
    int k = kc * 32 + kofs(e, lane);
    int col = lane & 15;
    float v = (k < H1) ? e_w2[(l * H1 + k) * Mm + col] : 0.f;
    frag[idx] = (_Float16)v;
}

// n_w1 [144,256] -> frags: [l][kchunk(5)][ntile(16)][lane*16+e]; kc=4 = m_i rows 128..143
__global__ void k_prep_nw1frag(const float* __restrict__ n_w1, _Float16* __restrict__ frag) {
    int idx = blockIdx.x * 256 + threadIdx.x;
    if (idx >= 4 * 5 * 16 * 512) return;
    int e = idx & 15, lane = (idx >> 4) & 31, t = idx >> 9;
    int nt = t % 16; t /= 16;
    int kc = t % 5;  int l = t / 5;
    int ko = kofs(e, lane);
    int row = (kc < 4) ? (kc * 32 + ko) : (ko < 16 ? 128 + ko : -1);
    int col = nt * 16 + (lane & 15);
    float v = (row >= 0) ? n_w1[(l * 144 + row) * 256 + col] : 0.f;
    frag[idx] = (_Float16)v;
}

// n_w2 [256,128] -> frags: [l][kchunk(8)][ntile(8)][lane*16+e]
__global__ void k_prep_nw2frag(const float* __restrict__ n_w2, _Float16* __restrict__ frag) {
    int idx = blockIdx.x * 256 + threadIdx.x;
    if (idx >= 4 * 8 * 8 * 512) return;
    int e = idx & 15, lane = (idx >> 4) & 31, t = idx >> 9;
    int nt = t % 8; t /= 8;
    int kc = t % 8; int l = t / 8;
    int row = kc * 32 + kofs(e, lane);
    int col = nt * 16 + (lane & 15);
    frag[idx] = (_Float16)n_w2[(l * 256 + row) * 128 + col];
}

// wd (dist row of e_w1) and e_b1, both zero-padded to KP
__global__ void k_prep_misc(const float* __restrict__ e_w1, const float* __restrict__ e_b1,
                            float* __restrict__ wdpad, float* __restrict__ b1pad) {
    int t = blockIdx.x * 256 + threadIdx.x;
    if (t >= 4 * KP) return;
    int l = t / KP, k = t % KP;
    wdpad[t] = (k < H1) ? e_w1[(l * EIN + 256) * H1 + k] : 0.f;
    b1pad[t] = (k < H1) ? e_b1[l * H1 + k] : 0.f;
}

// ---------------- gather / init ----------------

__global__ void k_gather(const int* __restrict__ z, const float* __restrict__ emb,
                         float* __restrict__ feats, _Float16* __restrict__ fbf) {
    int idx = blockIdx.x * 256 + threadIdx.x;   // ROWS*D
    int n = idx >> 7, d = idx & 127;
    float v = emb[z[n] * Dd + d];
    feats[idx] = v;
    fbf[idx] = (_Float16)v;
}

__global__ void k_coorinit(const float* __restrict__ pos, float* __restrict__ c0) {
    int idx = blockIdx.x * 256 + threadIdx.x;   // ROWS*3
    c0[idx] = pos[idx];
}

// ---------------- P/Q precompute: feats @ W1{a,b} via WMMA ----------------

__global__ void __launch_bounds__(256) k_pq(int l, const _Float16* __restrict__ fbf,
                                            const _Float16* __restrict__ w1frag,
                                            const float* __restrict__ b1pad,
                                            float* __restrict__ P, float* __restrict__ Q) {
    int wv = blockIdx.x * 8 + (threadIdx.x >> 5);   // 2*64*34 = 4352 wave-tiles
    int lane = threadIdx.x & 31;
    int nt = wv % NT1; wv /= NT1;
    int rt = wv % 64;  int which = wv / 64;         // 0=P, 1=Q
    int half8 = (lane & 16) ? 8 : 0;
    int row = rt * 16 + (lane & 15);
    const _Float16* frow = fbf + row * Dd;
    v8f c = {};
#pragma unroll
    for (int kc = 0; kc < 4; kc++) {
        int d0 = kc * 32;
        v8h lo = *(const v8h*)(frow + d0 + half8);
        v8h hi = *(const v8h*)(frow + d0 + 16 + half8);
        v16h a;
#pragma unroll
        for (int t = 0; t < 8; t++) { a[t] = lo[t]; a[t + 8] = hi[t]; }
        v16h b = *(const v16h*)(w1frag + ((((l * 2 + which) * 4 + kc) * NT1 + nt) * 512) + lane * 16);
        c = __builtin_amdgcn_wmma_f32_16x16x32_f16(false, a, false, b, (short)0, c, false, false);
    }
    int col = nt * 16 + (lane & 15);
    float* out = which ? Q : P;
    float bias = which ? 0.f : b1pad[l * KP + col];
#pragma unroll
    for (int r = 0; r < 8; r++) {
        int rg = rt * 16 + r + half8;
        out[rg * KP + col] = c[r] + bias;
    }
}

// ---------------- edge kernel: one (b,i) per workgroup, 8 waves cover all j ----------------

__global__ void __launch_bounds__(256) k_edge(int l, const float* __restrict__ coorsIn,
        float* __restrict__ coorsOut, const unsigned char* __restrict__ mask,
        const float* __restrict__ P, const float* __restrict__ Q,
        const float* __restrict__ wdpad, const _Float16* __restrict__ ew2frag,
        const float* __restrict__ e_b2, const float* __restrict__ c_w1,
        const float* __restrict__ c_b1, const float* __restrict__ c_w2,
        const float* __restrict__ c_b2, float* __restrict__ m_i) {
    __shared__ float s_cj[Nn * 3];
    __shared__ float s_mkf[Nn];
    __shared__ __align__(16) float s_P[KP];
    __shared__ __align__(16) float s_wd[KP];
    __shared__ float s_cw1[Mm * 64];
    __shared__ float s_cb1[64];
    __shared__ float s_cw2[64];
    __shared__ float s_eb2[Mm];
    __shared__ float s_macc[Mm];
    __shared__ float s_cdel[3];
    __shared__ float s_mt[8 * 256];

    int tid = threadIdx.x;
    int b = blockIdx.x >> 7;
    int i = blockIdx.x & 127;
    const float* cb = coorsIn + b * Nn * 3;
    for (int t = tid; t < Nn * 3; t += 256) s_cj[t] = cb[t];
    for (int t = tid; t < Nn; t += 256) s_mkf[t] = mask[b * Nn + t] ? 1.f : 0.f;
    const float* prow = P + (b * Nn + i) * KP;
    for (int t = tid; t < KP; t += 256) { s_P[t] = prow[t]; s_wd[t] = wdpad[l * KP + t]; }
    for (int t = tid; t < Mm * 64; t += 256) s_cw1[t] = c_w1[l * Mm * 64 + t];
    if (tid < 64) { s_cb1[tid] = c_b1[l * 64 + tid]; s_cw2[tid] = c_w2[l * 64 + tid]; }
    if (tid < 16) { s_eb2[tid] = e_b2[l * 16 + tid]; s_macc[tid] = 0.f; }
    if (tid < 3) s_cdel[tid] = 0.f;
    __syncthreads();

    int w = tid >> 5, lane = tid & 31;
    int j0 = w * 16;
    int jl = lane & 15;
    int jm = j0 + jl;
    int half8 = (lane & 16) ? 8 : 0;
    float cix = s_cj[i * 3], ciy = s_cj[i * 3 + 1], ciz = s_cj[i * 3 + 2];
    float dx = cix - s_cj[jm * 3], dy = ciy - s_cj[jm * 3 + 1], dz = ciz - s_cj[jm * 3 + 2];
    float dist = dx * dx + dy * dy + dz * dz;
    const float* qrow = Q + (b * Nn + jm) * KP;
    float mfi = s_mkf[i];

    v8f c = {};
    for (int kc = 0; kc < 17; kc++) {
        int k0 = kc * 32;
        v4f qa0 = *(const v4f*)(qrow + k0 + half8);
        v4f qa1 = *(const v4f*)(qrow + k0 + half8 + 4);
        v4f qb0 = *(const v4f*)(qrow + k0 + 16 + half8);
        v4f qb1 = *(const v4f*)(qrow + k0 + 16 + half8 + 4);
        v4f pa0 = *(const v4f*)(s_P + k0 + half8);
        v4f pa1 = *(const v4f*)(s_P + k0 + half8 + 4);
        v4f pb0 = *(const v4f*)(s_P + k0 + 16 + half8);
        v4f pb1 = *(const v4f*)(s_P + k0 + 16 + half8 + 4);
        v4f wa0 = *(const v4f*)(s_wd + k0 + half8);
        v4f wa1 = *(const v4f*)(s_wd + k0 + half8 + 4);
        v4f wb0 = *(const v4f*)(s_wd + k0 + 16 + half8);
        v4f wb1 = *(const v4f*)(s_wd + k0 + 16 + half8 + 4);
        v16h a;
#pragma unroll
        for (int t = 0; t < 4; t++) a[t]      = (_Float16)siluf(pa0[t] + qa0[t] + dist * wa0[t]);
#pragma unroll
        for (int t = 0; t < 4; t++) a[4 + t]  = (_Float16)siluf(pa1[t] + qa1[t] + dist * wa1[t]);
#pragma unroll
        for (int t = 0; t < 4; t++) a[8 + t]  = (_Float16)siluf(pb0[t] + qb0[t] + dist * wb0[t]);
#pragma unroll
        for (int t = 0; t < 4; t++) a[12 + t] = (_Float16)siluf(pb1[t] + qb1[t] + dist * wb1[t]);
        v16h bf = *(const v16h*)(ew2frag + (l * 17 + kc) * 512 + lane * 16);
        c = __builtin_amdgcn_wmma_f32_16x16x32_f16(false, a, false, bf, (short)0, c, false, false);
    }

    // m_ij = silu(C + b2): C layout -> row j = j0 + r + half8, col n = jl
    float eb = s_eb2[jl];
    float part = 0.f;
#pragma unroll
    for (int r = 0; r < 8; r++) {
        int rl = r + half8;
        float m = siluf(c[r] + eb);
        s_mt[w * 256 + rl * 16 + jl] = m;
        part += m * s_mkf[j0 + rl];
    }
    part *= mfi;
    part += __shfl_xor(part, 16);
    if (lane < 16) atomicAdd(&s_macc[jl], part);

    // coors MLP: lane L and L+16 split the 64 hidden units of row j = j0 + jl
    int hbase = (lane >= 16) ? 32 : 0;
    float mrow[16];
#pragma unroll
    for (int n = 0; n < 16; n++) mrow[n] = s_mt[w * 256 + jl * 16 + n];
    float s = 0.f;
    for (int hh = 0; hh < 32; hh++) {
        int h = hbase + hh;
        float acc = s_cb1[h];
#pragma unroll
        for (int n = 0; n < 16; n++) acc += mrow[n] * s_cw1[n * 64 + h];
        s += siluf(acc) * s_cw2[h];
    }
    s += __shfl_xor(s, 16);
    int jg = j0 + jl;
    float cw = (s + c_b2[l]) * mfi * s_mkf[jg];
    float gx = cw * (cix - s_cj[jg * 3]);
    float gy = cw * (ciy - s_cj[jg * 3 + 1]);
    float gz = cw * (ciz - s_cj[jg * 3 + 2]);
    if (lane >= 16) { gx = 0.f; gy = 0.f; gz = 0.f; }
#pragma unroll
    for (int o = 16; o >= 1; o >>= 1) {
        gx += __shfl_xor(gx, o); gy += __shfl_xor(gy, o); gz += __shfl_xor(gz, o);
    }
    if (lane == 0) {
        atomicAdd(&s_cdel[0], gx); atomicAdd(&s_cdel[1], gy); atomicAdd(&s_cdel[2], gz);
    }
    __syncthreads();
    if (tid < 16) m_i[(b * Nn + i) * Mm + tid] = s_macc[tid];
    if (tid < 3) coorsOut[(b * Nn + i) * 3 + tid] = s_cj[i * 3 + tid] + s_cdel[tid];
}

// ---------------- node kernel: LN + concat + MLP (two WMMA stages) ----------------

__global__ void __launch_bounds__(256) k_node(int l, float* __restrict__ feats,
        _Float16* __restrict__ fbf, const float* __restrict__ m_i,
        const float* __restrict__ ln_g, const float* __restrict__ ln_b,
        const _Float16* __restrict__ nw1frag, const float* __restrict__ n_b1,
        const _Float16* __restrict__ nw2frag, const float* __restrict__ n_b2) {
    __shared__ __align__(16) float s_f[16 * Dd];
    __shared__ __align__(16) _Float16 s_fn[16 * Dd];
    __shared__ __align__(16) _Float16 s_mh[16 * Mm];
    __shared__ __align__(16) _Float16 s_g1[16 * 256];
    __shared__ float s_red[512];

    int tid = threadIdx.x;
    int r0 = blockIdx.x * 16;
    int row = tid >> 4, sub = tid & 15;
    float x[8];
    float sm = 0.f, sq = 0.f;
    const float* fr = feats + (r0 + row) * Dd + sub * 8;
#pragma unroll
    for (int t = 0; t < 8; t++) {
        x[t] = fr[t]; sm += x[t]; sq += x[t] * x[t];
        s_f[row * Dd + sub * 8 + t] = x[t];
    }
    s_red[row * 16 + sub] = sm;
    s_red[256 + row * 16 + sub] = sq;
    s_mh[tid] = (_Float16)m_i[(r0 + row) * Mm + sub];
    __syncthreads();
    float mu = 0.f, vv = 0.f;
#pragma unroll
    for (int t = 0; t < 16; t++) { mu += s_red[row * 16 + t]; vv += s_red[256 + row * 16 + t]; }
    mu *= (1.f / 128.f);
    vv = vv * (1.f / 128.f) - mu * mu;
    float rs = rsqrtf(vv + 1e-5f);
#pragma unroll
    for (int t = 0; t < 8; t++) {
        int d = sub * 8 + t;
        s_fn[row * Dd + d] = (_Float16)((x[t] - mu) * rs * ln_g[l * Dd + d] + ln_b[l * Dd + d]);
    }
    __syncthreads();

    int w = tid >> 5, lane = tid & 31, half8 = (lane & 16) ? 8 : 0, rl = lane & 15;
    // GEMM1: [16 x 144pad160] @ [160 x 256]; wave covers ntiles w and w+8
    v8f c0 = {}, c1 = {};
#pragma unroll
    for (int kc = 0; kc < 5; kc++) {
        v16h a;
        if (kc < 4) {
            const _Float16* fp = s_fn + rl * Dd + kc * 32;
            v8h lo = *(const v8h*)(fp + half8);
            v8h hi = *(const v8h*)(fp + 16 + half8);
#pragma unroll
            for (int t = 0; t < 8; t++) { a[t] = lo[t]; a[t + 8] = hi[t]; }
        } else {
            v8h lo = *(const v8h*)(s_mh + rl * Mm + half8);
#pragma unroll
            for (int t = 0; t < 8; t++) { a[t] = lo[t]; a[t + 8] = (_Float16)0.f; }
        }
        const _Float16* bp = nw1frag + ((l * 5 + kc) * 16) * 512 + lane * 16;
        v16h b0 = *(const v16h*)(bp + w * 512);
        v16h b1 = *(const v16h*)(bp + (w + 8) * 512);
        c0 = __builtin_amdgcn_wmma_f32_16x16x32_f16(false, a, false, b0, (short)0, c0, false, false);
        c1 = __builtin_amdgcn_wmma_f32_16x16x32_f16(false, a, false, b1, (short)0, c1, false, false);
    }
#pragma unroll
    for (int r = 0; r < 8; r++) {
        int rr = r + half8;
        int col0 = w * 16 + rl, col1 = (w + 8) * 16 + rl;
        s_g1[rr * 256 + col0] = (_Float16)siluf(c0[r] + n_b1[l * 256 + col0]);
        s_g1[rr * 256 + col1] = (_Float16)siluf(c1[r] + n_b1[l * 256 + col1]);
    }
    __syncthreads();
    // GEMM2: [16 x 256] @ [256 x 128]; wave covers ntile w
    v8f c2 = {};
#pragma unroll
    for (int kc = 0; kc < 8; kc++) {
        const _Float16* gp = s_g1 + rl * 256 + kc * 32;
        v8h lo = *(const v8h*)(gp + half8);
        v8h hi = *(const v8h*)(gp + 16 + half8);
        v16h a;
#pragma unroll
        for (int t = 0; t < 8; t++) { a[t] = lo[t]; a[t + 8] = hi[t]; }
        v16h b = *(const v16h*)(nw2frag + ((l * 8 + kc) * 8 + w) * 512 + lane * 16);
        c2 = __builtin_amdgcn_wmma_f32_16x16x32_f16(false, a, false, b, (short)0, c2, false, false);
    }
    int col = w * 16 + rl;
#pragma unroll
    for (int r = 0; r < 8; r++) {
        int rr = r + half8;
        float v = s_f[rr * Dd + col] + c2[r] + n_b2[l * Dd + col];
        int g = (r0 + rr) * Dd + col;
        feats[g] = v;
        fbf[g] = (_Float16)v;
    }
}

// ---------------- masked mean pool ----------------

__global__ void k_pool(const float* __restrict__ feats, const unsigned char* __restrict__ mask,
                       float* __restrict__ out) {
    int b = blockIdx.x, d = threadIdx.x;
    float s = 0.f, wsum = 0.f;
    for (int n = 0; n < Nn; n++) {
        float mf = mask[b * Nn + n] ? 1.f : 0.f;
        s += feats[(b * Nn + n) * Dd + d] * mf;
        wsum += mf;
    }
    out[b * Dd + d] = s / fmaxf(wsum, 1.f);
}

// ---------------- host launch ----------------

extern "C" void kernel_launch(void* const* d_in, const int* in_sizes, int n_in,
                              void* d_out, int out_size, void* d_ws, size_t ws_size,
                              hipStream_t stream) {
    (void)in_sizes; (void)n_in; (void)out_size; (void)ws_size;
    const int*            z    = (const int*)d_in[0];
    const float*          pos  = (const float*)d_in[1];
    const unsigned char*  mask = (const unsigned char*)d_in[2];
    const float* emb  = (const float*)d_in[3];
    const float* e_w1 = (const float*)d_in[4];
    const float* e_b1 = (const float*)d_in[5];
    const float* e_w2 = (const float*)d_in[6];
    const float* e_b2 = (const float*)d_in[7];
    const float* c_w1 = (const float*)d_in[8];
    const float* c_b1 = (const float*)d_in[9];
    const float* c_w2 = (const float*)d_in[10];
    const float* c_b2 = (const float*)d_in[11];
    const float* ln_g = (const float*)d_in[12];
    const float* ln_b = (const float*)d_in[13];
    const float* n_w1 = (const float*)d_in[14];
    const float* n_b1 = (const float*)d_in[15];
    const float* n_w2 = (const float*)d_in[16];
    const float* n_b2 = (const float*)d_in[17];
    float* out = (float*)d_out;

    char* ws = (char*)d_ws;
    size_t off = 0;
    auto take = [&](size_t bytes) -> char* {
        char* p = ws + off;
        off += (bytes + 255) & ~(size_t)255;
        return p;
    };
    float*    feats   = (float*)   take(ROWS * Dd * 4);
    _Float16* fbf     = (_Float16*)take(ROWS * Dd * 2);
    float*    cbuf    = (float*)   take(2 * ROWS * 3 * 4);
    float*    P       = (float*)   take(ROWS * KP * 4);
    float*    Q       = (float*)   take(ROWS * KP * 4);
    float*    mi      = (float*)   take(ROWS * Mm * 4);
    _Float16* w1frag  = (_Float16*)take((size_t)4 * 2 * 4 * 34 * 512 * 2);
    _Float16* ew2frag = (_Float16*)take((size_t)4 * 17 * 512 * 2);
    _Float16* nw1frag = (_Float16*)take((size_t)4 * 5 * 16 * 512 * 2);
    _Float16* nw2frag = (_Float16*)take((size_t)4 * 8 * 8 * 512 * 2);
    float*    wdpad   = (float*)   take(4 * KP * 4);
    float*    b1pad   = (float*)   take(4 * KP * 4);

    k_prep_w1frag <<<2176, 256, 0, stream>>>(e_w1, w1frag);
    k_prep_ew2frag<<<136,  256, 0, stream>>>(e_w2, ew2frag);
    k_prep_nw1frag<<<640,  256, 0, stream>>>(n_w1, nw1frag);
    k_prep_nw2frag<<<512,  256, 0, stream>>>(n_w2, nw2frag);
    k_prep_misc   <<<9,    256, 0, stream>>>(e_w1, e_b1, wdpad, b1pad);
    k_gather      <<<512,  256, 0, stream>>>(z, emb, feats, fbf);
    k_coorinit    <<<12,   256, 0, stream>>>(pos, cbuf);

    for (int l = 0; l < DEPTH; l++) {
        float* cin  = cbuf + (l & 1) * ROWS * 3;
        float* cout = cbuf + ((l & 1) ^ 1) * ROWS * 3;
        k_pq  <<<544,  256, 0, stream>>>(l, fbf, w1frag, b1pad, P, Q);
        k_edge<<<1024, 256, 0, stream>>>(l, cin, cout, mask, P, Q, wdpad, ew2frag,
                                         e_b2, c_w1, c_b1, c_w2, c_b2, mi);
        k_node<<<64,   256, 0, stream>>>(l, feats, fbf, mi, ln_g, ln_b,
                                         nw1frag, n_b1, nw2frag, n_b2);
    }
    k_pool<<<Bsz, Dd, 0, stream>>>(feats, mask, out);
}
